// AnomalyDINOModel_58574763983226
// MI455X (gfx1250) — matrix operand order repaired and checked by reading, not compiled, not run on hardware
//
#include <hip/hip_runtime.h>

typedef _Float16 v16h __attribute__((ext_vector_type(16)));
typedef _Float16 v8h  __attribute__((ext_vector_type(8)));
typedef float    v8f  __attribute__((ext_vector_type(8)));
typedef unsigned int u32x4 __attribute__((ext_vector_type(4)));
typedef int          i32x8 __attribute__((ext_vector_type(8)));
typedef int          i32x4 __attribute__((ext_vector_type(4)));

#define D 384            // feature dim
#define NFEAT 8192       // 8*1024 feature rows
#define NBANK 65536      // memory bank rows
#define KCH 12           // 384 / 32 K-chunks
#define NSEG 8           // bank (N-dimension) segments
#define COLS 32          // bank columns staged per tile
#define NT 256           // tiles per segment: 65536/32/8
#define ROWB 768         // bytes per bank row in f16
#define TILE_LDS_BYTES 25600  // 32*768 + 48*16 pad + margin
#define IMGW 448
#define NPIX (8*448*448)

// TDM pads LDS with 16B every 512B stored; same mapping on the read side.
__device__ __forceinline__ unsigned pad_off(unsigned lin) {
  return lin + ((lin >> 9) << 4);
}

// ---------------- row L2-normalize fp32 -> f16, wave per row ----------------
__global__ __launch_bounds__(256) void rownorm_f16(const float* __restrict__ src,
                                                   _Float16* __restrict__ dst) {
  int lane = threadIdx.x & 31;
  int row  = blockIdx.x * 8 + (threadIdx.x >> 5);
  const float* r = src + (size_t)row * D;
  float v[12];
  float ss = 0.f;
#pragma unroll
  for (int j = 0; j < 12; ++j) { v[j] = r[lane + j * 32]; ss += v[j] * v[j]; }
#pragma unroll
  for (int m = 16; m >= 1; m >>= 1) ss += __shfl_xor(ss, m, 32);
  float inv = 1.0f / fmaxf(sqrtf(ss), 1e-12f);
  _Float16* o = dst + (size_t)row * D;
#pragma unroll
  for (int j = 0; j < 12; ++j) o[lane + j * 32] = (_Float16)(v[j] * inv);
}

#if __has_builtin(__builtin_amdgcn_tensor_load_to_lds)
#define USE_TDM 1
// Issue one TDM DMA: 32 rows x 384 f16 from global into LDS (padded layout).
// D# per CDNA5 ISA ch.8: group0 = {flags, lds_addr, global_addr lo, hi|type},
// group1 = {mask/size/pad, dims/tiles/strides}, groups 2-3 unused (2D tile).
// This toolchain exposes the 6-arg form: (g0, g1, g2, g3, g_extra8, cpol).
__device__ __forceinline__ void tdm_load_tile(const _Float16* gsrc, unsigned ldsoff) {
  unsigned long long ga = (unsigned long long)(const void*)gsrc;
  u32x4 g0;
  g0[0] = 1u;                                   // count=1, user mode
  g0[1] = ldsoff;                               // lds_addr (bytes)
  g0[2] = (unsigned)ga;                         // global_addr[31:0]
  g0[3] = ((unsigned)(ga >> 32) & 0x01FFFFFFu)  // global_addr[56:32]
          | 0x80000000u;                        // type=2 ("image")
  i32x8 g1;
  g1[0] = (int)((1u << 16)      // data_size = 2 bytes
              | (1u << 20)      // pad_enable
              | (6u << 22)      // pad_interval: every 512 B
              | (3u << 25));    // pad_amount: 4 dwords = 16 B
  g1[1] = (int)(384u << 16);            // tensor_dim0 = 384 (low 16 bits)
  g1[2] = 0;                            // dim0 hi = 0, dim1 lo16 = 0 (65536)
  g1[3] = (int)(1u | (384u << 16));     // dim1 hi = 1, tile_dim0 = 384
  g1[4] = (int)COLS;                    // tile_dim1 = 32 rows, tile_dim2 = 0
  g1[5] = 384;                          // tensor_dim0_stride = 384 elements
  g1[6] = 0;
  g1[7] = 0;
  i32x4 z4; z4[0] = 0; z4[1] = 0; z4[2] = 0; z4[3] = 0;
  i32x8 z8; z8[0] = 0; z8[1] = 0; z8[2] = 0; z8[3] = 0;
  z8[4] = 0; z8[5] = 0; z8[6] = 0; z8[7] = 0;
  __builtin_amdgcn_tensor_load_to_lds(g0, g1, z4, z4, z8, 0);
}
#else
#define USE_TDM 0
#endif

// ---------------- WMMA cosine-sim GEMM with running row-max ----------------
// grid.x = 64 row groups (128 rows, 8 waves x 16 rows)
// grid.y = 8 bank segments (256 32-col tiles each), double-buffered LDS + TDM
__global__ __launch_bounds__(256) void gemm_rowmax(const _Float16* __restrict__ Ah,
                                                   const _Float16* __restrict__ Bh,
                                                   float* __restrict__ pmax) {
  __shared__ __align__(128) char ldsB[2][TILE_LDS_BYTES];
  const int tid  = threadIdx.x;
  const int wave = tid >> 5;
  const int lane = tid & 31;
  const int m    = lane & 15;
  const int half = lane >> 4;
  const int rowbase = blockIdx.x * 128 + wave * 16;
  const int seg     = blockIdx.y;
  const int col0    = seg * NT * COLS;

  // A tile (16 rows x K=384) in registers, contiguous K layout (matches B):
  //  lane<16 : elems = K[c*32 + 0..15]; lane>=16: K[c*32 + 16..31]
  v16h a[KCH];
  {
    const _Float16* ap = Ah + (size_t)(rowbase + m) * D;
#pragma unroll
    for (int c = 0; c < KCH; ++c) {
      v8h lo = *(const v8h*)(ap + c * 32 + half * 16);
      v8h hi = *(const v8h*)(ap + c * 32 + half * 16 + 8);
      a[c] = __builtin_shufflevector(lo, hi, 0,1,2,3,4,5,6,7,8,9,10,11,12,13,14,15);
    }
  }

  float vmax[8];
#pragma unroll
  for (int i = 0; i < 8; ++i) vmax[i] = -3.0e38f;

  const unsigned lin0 = (unsigned)m * ROWB + half * 32;          // sub-tile 0 row
  const unsigned lin1 = (unsigned)(m + 16) * ROWB + half * 32;   // sub-tile 1 row

#if USE_TDM
  if (wave == 0)
    tdm_load_tile(Bh + (size_t)col0 * D, (unsigned)(size_t)&ldsB[0][0]);

  for (int t = 0; t < NT; ++t) {
    if (wave == 0) {
      if (t + 1 < NT) {
        tdm_load_tile(Bh + (size_t)(col0 + (t + 1) * COLS) * D,
                      (unsigned)(size_t)&ldsB[(t + 1) & 1][0]);
        __builtin_amdgcn_s_wait_tensorcnt((short)1);   // tile t landed
      } else {
        __builtin_amdgcn_s_wait_tensorcnt((short)0);
      }
    }
    __syncthreads();                                   // buf[t&1] ready for all
    const char* base = &ldsB[t & 1][0];
    v8f acc0 = {}, acc1 = {};
#pragma unroll
    for (int c = 0; c < KCH; ++c) {
      unsigned o0 = pad_off(lin0 + c * 64);
      v8h b0a = *(const v8h*)(base + o0);
      v8h b0b = *(const v8h*)(base + o0 + 16);
      v16h b0 = __builtin_shufflevector(b0a, b0b, 0,1,2,3,4,5,6,7,8,9,10,11,12,13,14,15);
      acc0 = __builtin_amdgcn_wmma_f32_16x16x32_f16(false, a[c], false, b0,
                                                    (short)0, acc0, false, false);
      unsigned o1 = pad_off(lin1 + c * 64);
      v8h b1a = *(const v8h*)(base + o1);
      v8h b1b = *(const v8h*)(base + o1 + 16);
      v16h b1 = __builtin_shufflevector(b1a, b1b, 0,1,2,3,4,5,6,7,8,9,10,11,12,13,14,15);
      acc1 = __builtin_amdgcn_wmma_f32_16x16x32_f16(false, a[c], false, b1,
                                                    (short)0, acc1, false, false);
    }
#pragma unroll
    for (int i = 0; i < 8; ++i)
      vmax[i] = fmaxf(vmax[i], fmaxf(acc0[i], acc1[i]));
    __syncthreads();                                   // done reading buf[t&1]
  }
#else
  // Fallback: manual double-buffered copy (loads overlap compute).
  {
    char* b0 = &ldsB[0][0];
#pragma unroll
    for (int i = 0; i < 6; ++i) {
      int j = tid + i * 256, rr = j / 48, q = j % 48;
      unsigned l = (unsigned)j * 16;
      *(uint4*)(b0 + pad_off(l)) =
          ((const uint4*)(Bh + (size_t)(col0 + rr) * D))[q];
    }
  }
  __syncthreads();
  for (int t = 0; t < NT; ++t) {
    uint4 tmp[6];
    if (t + 1 < NT) {
      const int n0n = col0 + (t + 1) * COLS;
#pragma unroll
      for (int i = 0; i < 6; ++i) {
        int j = tid + i * 256, rr = j / 48, q = j % 48;
        tmp[i] = ((const uint4*)(Bh + (size_t)(n0n + rr) * D))[q];
      }
    }
    const char* base = &ldsB[t & 1][0];
    v8f acc0 = {}, acc1 = {};
#pragma unroll
    for (int c = 0; c < KCH; ++c) {
      unsigned o0 = pad_off(lin0 + c * 64);
      v8h b0a = *(const v8h*)(base + o0);
      v8h b0b = *(const v8h*)(base + o0 + 16);
      v16h b0 = __builtin_shufflevector(b0a, b0b, 0,1,2,3,4,5,6,7,8,9,10,11,12,13,14,15);
      acc0 = __builtin_amdgcn_wmma_f32_16x16x32_f16(false, a[c], false, b0,
                                                    (short)0, acc0, false, false);
      unsigned o1 = pad_off(lin1 + c * 64);
      v8h b1a = *(const v8h*)(base + o1);
      v8h b1b = *(const v8h*)(base + o1 + 16);
      v16h b1 = __builtin_shufflevector(b1a, b1b, 0,1,2,3,4,5,6,7,8,9,10,11,12,13,14,15);
      acc1 = __builtin_amdgcn_wmma_f32_16x16x32_f16(false, a[c], false, b1,
                                                    (short)0, acc1, false, false);
    }
#pragma unroll
    for (int i = 0; i < 8; ++i)
      vmax[i] = fmaxf(vmax[i], fmaxf(acc0[i], acc1[i]));
    if (t + 1 < NT) {
      char* bn = &ldsB[(t + 1) & 1][0];
#pragma unroll
      for (int i = 0; i < 6; ++i) {
        int j = tid + i * 256;
        unsigned l = (unsigned)j * 16;
        *(uint4*)(bn + pad_off(l)) = tmp[i];
      }
    }
    __syncthreads();
  }
#endif

  // C layout: VGPR i -> row i (lanes 0-15) / row 8+i (lanes 16-31); col = lane&15.
#pragma unroll
  for (int i = 0; i < 8; ++i) {
    float v = vmax[i];
#pragma unroll
    for (int msk = 8; msk >= 1; msk >>= 1) v = fmaxf(v, __shfl_xor(v, msk, 32));
    if (m == 0)
      pmax[(size_t)(rowbase + i + half * 8) * NSEG + seg] = v;
  }
}

// ---------------- combine segment maxima -> min distance ----------------
__global__ void reduce_min_d(const float* __restrict__ pmax, float* __restrict__ min_d) {
  int row = blockIdx.x * 256 + threadIdx.x;
  if (row >= NFEAT) return;
  float v = -3.0e38f;
#pragma unroll
  for (int s = 0; s < NSEG; ++s) v = fmaxf(v, pmax[(size_t)row * NSEG + s]);
  float d = 1.0f - v;
  min_d[row] = fminf(fmaxf(d, 0.0f), 2.0f);
}

// ---------------- top-10 mean per batch ----------------
__global__ __launch_bounds__(256) void topk_mean(const float* __restrict__ min_d,
                                                 float* __restrict__ score) {
  __shared__ float vals[1024];
  __shared__ float sv[256];
  __shared__ int   si[256];
  const int b = blockIdx.x, tid = threadIdx.x;
  for (int j = tid; j < 1024; j += 256) vals[j] = min_d[b * 1024 + j];
  __syncthreads();
  float sum = 0.f;
  for (int it = 0; it < 10; ++it) {
    float best = -1e30f; int bi = 0;
    for (int j = tid; j < 1024; j += 256) {
      float x = vals[j];
      if (x > best) { best = x; bi = j; }
    }
    sv[tid] = best; si[tid] = bi;
    __syncthreads();
    for (int s = 128; s > 0; s >>= 1) {
      if (tid < s && sv[tid + s] > sv[tid]) { sv[tid] = sv[tid + s]; si[tid] = si[tid + s]; }
      __syncthreads();
    }
    if (tid == 0) { sum += sv[0]; vals[si[0]] = -1e30f; }
    __syncthreads();
  }
  if (tid == 0) score[b] = sum * 0.1f;
}

// ---------------- bilinear 32x32 -> 448x448 (half-pixel, clamped) ----------------
__global__ void bilinear_up(const float* __restrict__ min_d, float* __restrict__ out) {
  int idx = blockIdx.x * 256 + threadIdx.x;
  if (idx >= NPIX) return;
  int x = idx % IMGW;
  int y = (idx / IMGW) % IMGW;
  int b = idx / (IMGW * IMGW);
  float sx = fminf(fmaxf((x + 0.5f) * (1.0f / 14.0f) - 0.5f, 0.0f), 31.0f);
  float sy = fminf(fmaxf((y + 0.5f) * (1.0f / 14.0f) - 0.5f, 0.0f), 31.0f);
  int x0 = (int)sx, y0 = (int)sy;
  int x1 = x0 + 1 > 31 ? 31 : x0 + 1;
  int y1 = y0 + 1 > 31 ? 31 : y0 + 1;
  float fx = sx - x0, fy = sy - y0;
  const float* g = min_d + b * 1024;
  float v00 = g[y0 * 32 + x0], v01 = g[y0 * 32 + x1];
  float v10 = g[y1 * 32 + x0], v11 = g[y1 * 32 + x1];
  out[idx] = (1.f - fy) * ((1.f - fx) * v00 + fx * v01)
           +        fy  * ((1.f - fx) * v10 + fx * v11);
}

// ---------------- separable Gaussian blur, sigma=4, radius=16, reflect pad ----------------
__device__ __forceinline__ void make_gauss(float* kw, int tid) {
  if (tid == 0) {
    float tmp[33], s = 0.f;
    for (int t = 0; t < 33; ++t) {
      float u = (t - 16) * 0.25f;
      tmp[t] = expf(-0.5f * u * u);
      s += tmp[t];
    }
    float inv = 1.0f / s;
    for (int t = 0; t < 33; ++t) kw[t] = tmp[t] * inv;
  }
}

__global__ void blur_h(const float* __restrict__ in, float* __restrict__ out) {
  __shared__ float kw[33];
  make_gauss(kw, threadIdx.x);
  __syncthreads();
  int idx = blockIdx.x * 256 + threadIdx.x;
  if (idx >= NPIX) return;
  int x = idx % IMGW;
  const float* rowp = in + (idx / IMGW) * IMGW;
  float s = 0.f;
  for (int t = -16; t <= 16; ++t) {
    int xx = x + t;
    xx = xx < 0 ? -xx : (xx > 447 ? 894 - xx : xx);
    s += kw[t + 16] * rowp[xx];
  }
  out[idx] = s;
}

__global__ void blur_v(const float* __restrict__ in, float* __restrict__ out) {
  __shared__ float kw[33];
  make_gauss(kw, threadIdx.x);
  __syncthreads();
  int idx = blockIdx.x * 256 + threadIdx.x;
  if (idx >= NPIX) return;
  int x = idx % IMGW;
  int y = (idx / IMGW) % IMGW;
  int b = idx / (IMGW * IMGW);
  const float* img = in + b * (IMGW * IMGW);
  float s = 0.f;
  for (int t = -16; t <= 16; ++t) {
    int yy = y + t;
    yy = yy < 0 ? -yy : (yy > 447 ? 894 - yy : yy);
    s += kw[t + 16] * img[yy * IMGW + x];
  }
  out[idx] = s;
}

extern "C" void kernel_launch(void* const* d_in, const int* in_sizes, int n_in,
                              void* d_out, int out_size, void* d_ws, size_t ws_size,
                              hipStream_t stream) {
  const float* features = (const float*)d_in[0];   // (8,1024,384) f32
  const float* bank     = (const float*)d_in[1];   // (65536,384)  f32
  float* out = (float*)d_out;                      // [8 scores][8*448*448 map]

  char* w = (char*)d_ws;
  _Float16* Ah   = (_Float16*)(w);                                   //  6,291,456 B
  _Float16* Bh   = (_Float16*)(w + 6291456);                         // 50,331,648 B
  float*    pmax = (float*)  (w + 6291456 + 50331648);               //    262,144 B
  float*    mind = (float*)  (w + 6291456 + 50331648 + 262144);      //     32,768 B
  float*    T0   = (float*)  (w + 6291456 + 50331648 + 262144 + 32768);            // 6,422,528 B
  float*    T1   = (float*)  (w + 6291456 + 50331648 + 262144 + 32768 + 6422528);  // 6,422,528 B

  rownorm_f16<<<NFEAT / 8, 256, 0, stream>>>(features, Ah);
  rownorm_f16<<<NBANK / 8, 256, 0, stream>>>(bank, Bh);

  gemm_rowmax<<<dim3(NFEAT / 128, NSEG), 256, 0, stream>>>(Ah, Bh, pmax);

  reduce_min_d<<<(NFEAT + 255) / 256, 256, 0, stream>>>(pmax, mind);
  topk_mean<<<8, 256, 0, stream>>>(mind, out);

  const int nblk = (NPIX + 255) / 256;
  bilinear_up<<<nblk, 256, 0, stream>>>(mind, T0);
  blur_h<<<nblk, 256, 0, stream>>>(T0, T1);
  blur_v<<<nblk, 256, 0, stream>>>(T1, out + 8);
}
